// MHAttention_19756849561631
// MI455X (gfx1250) — matrix operand order repaired and checked by reading, not compiled, not run on hardware
//
#include <hip/hip_runtime.h>

// ---------------------------------------------------------------------------
// MHA forward for MI455X (gfx1250, wave32, WMMA).
// - All GEMMs: v_wmma_f32_16x16x32_bf16 (f32 accumulate)
// - Attention: flash-style online softmax, logits never stored
// - kv tiles staged into LDS via global_load_async_to_lds_b128 (ASYNCcnt)
// ---------------------------------------------------------------------------

typedef __attribute__((ext_vector_type(16))) __bf16 v16bf;
typedef __attribute__((ext_vector_type(8)))  __bf16 v8bf;
typedef __attribute__((ext_vector_type(8)))  float  v8f;

constexpr int kB    = 2;
constexpr int kN    = 2048;
constexpr int kDim  = 1024;
constexpr int kH    = 16;
constexpr int kDH   = 64;
constexpr int kBH   = kB * kH;       // 32
constexpr int kM    = kB * kN;       // 4096 rows of the flattened GEMMs
constexpr float kScale = 0.125f;     // DH^-0.5
constexpr float kLn1e4 = 9.2103403719761836f;

static __device__ __forceinline__ __bf16 f2bf(float f) {
  unsigned u = __float_as_uint(f);
  u += 0x7FFFu + ((u >> 16) & 1u);            // round-to-nearest-even
  unsigned short h = (unsigned short)(u >> 16);
  return __builtin_bit_cast(__bf16, h);
}

static __device__ __forceinline__ v8f wmma_bf16(v16bf a, v16bf b, v8f c) {
  // (neg_a, A, neg_b, B, c_mod, C, reuse_a, reuse_b)
  return __builtin_amdgcn_wmma_f32_16x16x32_bf16(false, a, false, b,
                                                 (short)0, c, false, false);
}

// Async copy of one 16-byte chunk global -> LDS, tracked by ASYNCcnt.
// lds_off is the wave-relative LDS byte address (= low 32 bits of the
// generic pointer, per the LDS aperture mapping).
static __device__ __forceinline__ void async_copy_b128(unsigned lds_off,
                                                       const __bf16* gptr) {
  asm volatile("global_load_async_to_lds_b128 %0, %1, off"
               :: "v"(lds_off), "v"(gptr) : "memory");
}
static __device__ __forceinline__ void wait_asynccnt0() {
  asm volatile("s_wait_asynccnt 0x0" ::: "memory");
}

// A-matrix 16x32 bf16 fragment (ISA 7.12.2): lane lr holds row M=lr;
// half lh=0: K = [0..7] and [16..23]; lh=1: K = [8..15] and [24..31].
static __device__ __forceinline__ v16bf
load_fragA(const __bf16* base, size_t row, int ld, int kk, int lh) {
  const v8bf a = *reinterpret_cast<const v8bf*>(base + row * (size_t)ld + kk + lh * 8);
  const v8bf b = *reinterpret_cast<const v8bf*>(base + row * (size_t)ld + kk + 16 + lh * 8);
  v16bf r;
#pragma unroll
  for (int i = 0; i < 8; ++i) { r[i] = a[i]; r[i + 8] = b[i]; }
  return r;
}

// B-matrix 32x16 bf16 fragment: lane lr holds column N=lr;
// half lh=0: K = [0..7] and [16..23]; lh=1: K = [8..15] and [24..31].
static __device__ __forceinline__ v16bf
load_fragB(const __bf16* base, size_t col, int ld, int kk, int lh) {
  const v8bf* p = reinterpret_cast<const v8bf*>(base + col * (size_t)ld + kk + lh * 8);
  const v8bf a = p[0], b = p[2];
  v16bf r;
#pragma unroll
  for (int i = 0; i < 8; ++i) { r[i] = a[i]; r[i + 8] = b[i]; }
  return r;
}

// -------------------------- precision staging ------------------------------

__global__ __launch_bounds__(256) void cvt_f32_bf16_kernel(
    const float* __restrict__ s, __bf16* __restrict__ d, int n) {
  int i = blockIdx.x * 256 + threadIdx.x;
  if (i < n) d[i] = f2bf(s[i]);
}

// dst[c][r] = src[r][c]  (weights go K-major so B-fragments load contiguously)
__global__ __launch_bounds__(256) void transpose_cvt_kernel(
    const float* __restrict__ s, __bf16* __restrict__ d, int rows, int cols) {
  int i = blockIdx.x * 256 + threadIdx.x;
  if (i < rows * cols) {
    int r = i / cols, c = i % cols;
    d[(size_t)c * rows + r] = f2bf(s[i]);
  }
}

// ---------------- fused QKV projection + rotary + scale --------------------
// grid = (kM/128, kH); 8 waves/block, each wave: 16 rows x one head (64 cols).
// Writes qh/kvh as [bh][n][d] bf16 and kvhT as [bh][d][n] bf16.

__global__ __launch_bounds__(256) void proj_rope_kernel(
    const __bf16* __restrict__ xb, const __bf16* __restrict__ WqT,
    const __bf16* __restrict__ WkvT, __bf16* __restrict__ qh,
    __bf16* __restrict__ kvh, __bf16* __restrict__ kvhT) {
  const int lane = threadIdx.x & 31, wave = threadIdx.x >> 5;
  const int lr = lane & 15, lh = lane >> 4;
  const int m0 = blockIdx.x * 128 + wave * 16;
  const int h  = blockIdx.y;
  const int c0 = h * kDH;

  v8f accq[4] = {}; v8f acck[4] = {};
  for (int kk = 0; kk < kDim; kk += 32) {
    v16bf af = load_fragA(xb, (size_t)(m0 + lr), kDim, kk, lh);
#pragma unroll
    for (int t = 0; t < 4; ++t) {
      v16bf bq = load_fragB(WqT,  (size_t)(c0 + t * 16 + lr), kDim, kk, lh);
      accq[t] = wmma_bf16(af, bq, accq[t]);
      v16bf bk = load_fragB(WkvT, (size_t)(c0 + t * 16 + lr), kDim, kk, lh);
      acck[t] = wmma_bf16(af, bk, acck[t]);
    }
  }

  // inv_freq depends only on (t, lane): hoist out of the row loop.
  float invf[2];
#pragma unroll
  for (int t = 0; t < 2; ++t)
    invf[t] = __expf(-(float)(2 * (t * 16 + lr)) * (kLn1e4 / (float)kDH));

  // C layout: element (vgpr r, lane) = C[row r + 8*lh][col lr].
  // Tiles (t, t+2) pair head-dims d and d+32 for rotate_half.
#pragma unroll
  for (int r = 0; r < 8; ++r) {
    int m = m0 + r + 8 * lh;
    int b = m >> 11, n = m & (kN - 1);
    size_t bh = (size_t)(b * kH + h);
#pragma unroll
    for (int t = 0; t < 2; ++t) {
      int dlo = t * 16 + lr, dhi = dlo + 32;
      float f = (float)n * invf[t];
      float cf = __cosf(f), sf = __sinf(f);
      float qlo = accq[t][r], qhi = accq[t + 2][r];
      float klo = acck[t][r], khi = acck[t + 2][r];
      float qnl = (qlo * cf - qhi * sf) * kScale;
      float qnh = (qhi * cf + qlo * sf) * kScale;
      float knl = klo * cf - khi * sf;
      float knh = khi * cf + klo * sf;
      size_t rowb = (bh * kN + n) * kDH;
      qh [rowb + dlo] = f2bf(qnl);  qh [rowb + dhi] = f2bf(qnh);
      kvh[rowb + dlo] = f2bf(knl);  kvh[rowb + dhi] = f2bf(knh);
      size_t tb = bh * (size_t)kDH * kN + n;
      kvhT[tb + (size_t)dlo * kN] = f2bf(knl);
      kvhT[tb + (size_t)dhi * kN] = f2bf(knh);
    }
  }
}

// ------------------------- flash attention ---------------------------------
// grid = (kBH, kN/128); 8 waves/block. Wave owns 16 q-rows; the block streams
// kv in 32-row tiles. kv and kv^T tiles are staged into LDS once per block
// with global_load_async_to_lds_b128 (each thread moves one 16B chunk of
// each tile), then all 8 waves read their WMMA B-fragments from LDS.

__global__ __launch_bounds__(256) void flash_attn_kernel(
    const __bf16* __restrict__ qh, const __bf16* __restrict__ kvh,
    const __bf16* __restrict__ kvhT, __bf16* __restrict__ ctx) {
  __shared__ __align__(16) __bf16 kvs [32 * kDH];   // kv tile   [j][d]  4 KB
  __shared__ __align__(16) __bf16 kvts[kDH * 32];   // kv^T tile [d][j]  4 KB
  __shared__ __align__(16) __bf16 Pbuf[8][16][32];  //                   8 KB
  const int tid  = threadIdx.x;
  const int lane = tid & 31, wave = tid >> 5;
  const int lr = lane & 15, lh = lane >> 4;
  const int bh = blockIdx.x;
  const int b = bh >> 4, h = bh & 15;
  const int m0 = blockIdx.y * 128 + wave * 16;

  const __bf16* qbh = qh   + (size_t)bh * kN * kDH;
  const __bf16* kbh = kvh  + (size_t)bh * kN * kDH;
  const __bf16* ktb = kvhT + (size_t)bh * kDH * kN;

  // Per-thread staging assignment: one 16B chunk of each LDS tile.
  const int srow = tid >> 3, schk = tid & 7;   // kv tile:   j 0..31, 8 chunks/row
  const int trow = tid >> 2, tchk = tid & 3;   // kvT tile:  d 0..63, 4 chunks/row
  const unsigned kvs_off  = (unsigned)(uintptr_t)&kvs [srow * kDH + schk * 8];
  const unsigned kvts_off = (unsigned)(uintptr_t)&kvts[trow * 32  + tchk * 8];
  const __bf16* kv_g  = kbh + (size_t)srow * kDH + schk * 8;  // + jb*kDH per iter
  const __bf16* kvt_g = ktb + (size_t)trow * kN  + tchk * 8;  // + jb     per iter

  v16bf aq0 = load_fragA(qbh, (size_t)(m0 + lr), kDH, 0,  lh);
  v16bf aq1 = load_fragA(qbh, (size_t)(m0 + lr), kDH, 32, lh);

  v8f o[4] = {};
  float mrow[8], lrow[8];
#pragma unroll
  for (int r = 0; r < 8; ++r) { mrow[r] = -3.0e38f; lrow[r] = 0.0f; }

  for (int jb = 0; jb < kN; jb += 32) {
    __syncthreads();                       // previous tile fully consumed
    async_copy_b128(kvs_off,  kv_g  + (size_t)jb * kDH);
    async_copy_b128(kvts_off, kvt_g + jb);
    wait_asynccnt0();
    __syncthreads();                       // tile visible to all waves

    // S block 16x32 = q(16x64) @ kv^T : two 16x16 C tiles, K=64 in 2 steps.
    v8f s0 = {}, s1 = {};
    {
      v16bf b00 = load_fragB(kvs, (size_t)lr,        kDH, 0,  lh);
      v16bf b01 = load_fragB(kvs, (size_t)lr,        kDH, 32, lh);
      s0 = wmma_bf16(aq0, b00, s0);
      s0 = wmma_bf16(aq1, b01, s0);
      v16bf b10 = load_fragB(kvs, (size_t)(16 + lr), kDH, 0,  lh);
      v16bf b11 = load_fragB(kvs, (size_t)(16 + lr), kDH, 32, lh);
      s1 = wmma_bf16(aq0, b10, s1);
      s1 = wmma_bf16(aq1, b11, s1);
    }

    // Online softmax: rows live across the 16 lanes of a half-wave group.
    float osc[8];
#pragma unroll
    for (int r = 0; r < 8; ++r) {
      float v0 = s0[r], v1 = s1[r];
      float tm = fmaxf(v0, v1);
#pragma unroll
      for (int msk = 1; msk < 16; msk <<= 1)
        tm = fmaxf(tm, __shfl_xor(tm, msk, 32));
      float mnew = fmaxf(mrow[r], tm);
      float sc = __expf(mrow[r] - mnew);
      float p0 = __expf(v0 - mnew), p1 = __expf(v1 - mnew);
      float rs = p0 + p1;
#pragma unroll
      for (int msk = 1; msk < 16; msk <<= 1)
        rs += __shfl_xor(rs, msk, 32);
      lrow[r] = lrow[r] * sc + rs;
      mrow[r] = mnew;
      osc[r] = sc;
      Pbuf[wave][r + 8 * lh][lr]      = f2bf(p0);
      Pbuf[wave][r + 8 * lh][16 + lr] = f2bf(p1);
    }
#pragma unroll
    for (int t = 0; t < 4; ++t)
#pragma unroll
      for (int r = 0; r < 8; ++r) o[t][r] *= osc[r];

    // O(16x64) += P(16x32) @ KV(32x64); P re-read in A-fragment layout,
    // KV B-fragments are K-contiguous in the staged kv^T LDS tile.
    v16bf pf = load_fragA(&Pbuf[wave][0][0], (size_t)lr, 32, 0, lh);
#pragma unroll
    for (int t = 0; t < 4; ++t) {
      v16bf bk = load_fragB(kvts, (size_t)(t * 16 + lr), 32, 0, lh);
      o[t] = wmma_bf16(pf, bk, o[t]);
    }
  }

  // normalize and write merged-head context [b][n][h*64+d] as bf16
#pragma unroll
  for (int r = 0; r < 8; ++r) {
    int n = m0 + r + 8 * lh;
    float inv = 1.0f / lrow[r];
    size_t base = ((size_t)b * kN + n) * kDim + h * kDH;
#pragma unroll
    for (int t = 0; t < 4; ++t)
      ctx[base + t * 16 + lr] = f2bf(o[t][r] * inv);
  }
}

// ------------------------ output projection --------------------------------
// grid = (kM/128, kDim/64): ctx(bf16) @ WoutT + bout -> f32 output.

__global__ __launch_bounds__(256) void out_proj_kernel(
    const __bf16* __restrict__ ctx, const __bf16* __restrict__ WoT,
    const float* __restrict__ bout, float* __restrict__ out) {
  const int lane = threadIdx.x & 31, wave = threadIdx.x >> 5;
  const int lr = lane & 15, lh = lane >> 4;
  const int m0 = blockIdx.x * 128 + wave * 16;
  const int c0 = blockIdx.y * 64;

  v8f acc[4] = {};
  for (int kk = 0; kk < kDim; kk += 32) {
    v16bf af = load_fragA(ctx, (size_t)(m0 + lr), kDim, kk, lh);
#pragma unroll
    for (int t = 0; t < 4; ++t) {
      v16bf bw = load_fragB(WoT, (size_t)(c0 + t * 16 + lr), kDim, kk, lh);
      acc[t] = wmma_bf16(af, bw, acc[t]);
    }
  }
#pragma unroll
  for (int r = 0; r < 8; ++r) {
    int m = m0 + r + 8 * lh;
#pragma unroll
    for (int t = 0; t < 4; ++t) {
      int c = c0 + t * 16 + lr;
      out[(size_t)m * kDim + c] = acc[t][r] + bout[c];
    }
  }
}

// ---------------------------------------------------------------------------

extern "C" void kernel_launch(void* const* d_in, const int* in_sizes, int n_in,
                              void* d_out, int out_size, void* d_ws, size_t ws_size,
                              hipStream_t stream) {
  const float* x    = (const float*)d_in[0];
  const float* Wq   = (const float*)d_in[1];
  const float* Wkv  = (const float*)d_in[2];
  const float* Wout = (const float*)d_in[3];
  const float* bout = (const float*)d_in[4];
  float* out = (float*)d_out;

  char* ws = (char*)d_ws;
  const size_t MB = (size_t)1 << 20;
  __bf16* xb   = (__bf16*)(ws + 0  * MB);  // 8 MiB  x as bf16
  __bf16* WqT  = (__bf16*)(ws + 8  * MB);  // 2 MiB
  __bf16* WkvT = (__bf16*)(ws + 10 * MB);  // 2 MiB
  __bf16* WoT  = (__bf16*)(ws + 12 * MB);  // 2 MiB
  __bf16* qh   = (__bf16*)(ws + 14 * MB);  // 8 MiB  [bh][n][d]
  __bf16* kvh  = (__bf16*)(ws + 22 * MB);  // 8 MiB  [bh][n][d]
  __bf16* kvhT = (__bf16*)(ws + 30 * MB);  // 8 MiB  [bh][d][n]
  __bf16* ctx  = (__bf16*)(ws + 38 * MB);  // 8 MiB  [b][n][dim]

  const int nx = kM * kDim;
  cvt_f32_bf16_kernel<<<(nx + 255) / 256, 256, 0, stream>>>(x, xb, nx);
  const int nw = kDim * kDim;
  transpose_cvt_kernel<<<(nw + 255) / 256, 256, 0, stream>>>(Wq,   WqT,  kDim, kDim);
  transpose_cvt_kernel<<<(nw + 255) / 256, 256, 0, stream>>>(Wkv,  WkvT, kDim, kDim);
  transpose_cvt_kernel<<<(nw + 255) / 256, 256, 0, stream>>>(Wout, WoT,  kDim, kDim);

  proj_rope_kernel<<<dim3(kM / 128, kH), 256, 0, stream>>>(xb, WqT, WkvT, qh, kvh, kvhT);
  flash_attn_kernel<<<dim3(kBH, kN / 128), 256, 0, stream>>>(qh, kvh, kvhT, ctx);
  out_proj_kernel<<<dim3(kM / 128, kDim / 64), 256, 0, stream>>>(ctx, WoT, bout, out);
}
